// PoseCameraCrossAttention_62491774157537
// MI455X (gfx1250) — matrix operand order repaired and checked by reading, not compiled, not run on hardware
//
#include <hip/hip_runtime.h>
#include <hip/hip_bf16.h>

// ---------------------------------------------------------------------------
// PoseCameraCrossAttention for MI455X (gfx1250, wave32, WMMA bf16 + TDM).
// B=8 T=32 K=8 P=4 J=24 D=256 H=8 dh=32.
// ---------------------------------------------------------------------------

typedef __attribute__((ext_vector_type(16))) __bf16 v16bf;
typedef __attribute__((ext_vector_type(8)))  float  v8f;
typedef __attribute__((ext_vector_type(4)))  unsigned int u32x4;
typedef __attribute__((ext_vector_type(8)))  int i32x8;
typedef __attribute__((ext_vector_type(4)))  int i32x4;

union V16U { uint4 q[2]; v16bf v; };

#define POSE_ELEMS  50331648   // 8*32*8*4*24*256
#define CAM_ELEMS   524288     // 8*32*8*256
#define WBF_FOFF    1572864    // float offset of bf16 weight area in ws
#define ATT_SCALE   0.17677669529663687f   // 1/sqrt(32)

__device__ __forceinline__ float wave_sum(float v) {
#pragma unroll
  for (int off = 16; off > 0; off >>= 1) v += __shfl_xor(v, off, 32);
  return v;
}
__device__ __forceinline__ float wave_max(float v) {
#pragma unroll
  for (int off = 16; off > 0; off >>= 1) v = fmaxf(v, __shfl_xor(v, off, 32));
  return v;
}

// ---------------------------------------------------------------------------
// Tensor Data Mover: async-copy 128 KB (32768 dwords) from global to LDS.
// D# groups per CDNA5 ISA ch.8: 1 valid descriptor, 1-row tile of 32768
// 4-byte elements, no padding/iterate/gather, no multicast.
// ---------------------------------------------------------------------------
__device__ __forceinline__ void tdm_load_128k(const void* gsrc, unsigned lds_byte_off) {
  unsigned long long ga = (unsigned long long)gsrc;
  u32x4 g0;
  g0[0] = 1u;                                    // count=1, is_restore=0
  g0[1] = lds_byte_off;                          // lds_addr
  g0[2] = (unsigned)(ga & 0xffffffffu);          // global_addr[31:0]
  g0[3] = (unsigned)(ga >> 32) | 0x80000000u;    // global_addr[56:32] | type=2
  i32x8 g1;
  g1[0] = 0x20000;            // data_size=2 (4 bytes), mask=0
  g1[1] = (int)0x80000000u;   // tensor_dim0[15:0]=0x8000 in bits[31:16]
  g1[2] = 0x10000;            // tensor_dim0[31:16]=0 ; tensor_dim1[15:0]=1
  g1[3] = (int)0x80000000u;   // tensor_dim1[31:16]=0 ; tile_dim0=0x8000
  g1[4] = 1;                  // tile_dim1=1, tile_dim2=0
  g1[5] = 32768;              // tensor_dim0_stride[31:0]
  g1[6] = 0;                  // stride hi / tensor_dim1_stride lo
  g1[7] = 0;
  i32x4 z4 = {};
#if __clang_major__ >= 23
  i32x8 z8 = {};
  __builtin_amdgcn_tensor_load_to_lds(g0, g1, z4, z4, z8, 0);
#else
  __builtin_amdgcn_tensor_load_to_lds(g0, g1, z4, z4, 0);
#endif
}

// A fragment (16x32 bf16) from an LDS matrix with row stride 256 elements.
__device__ __forceinline__ v16bf a_frag(const __bf16* mat, int m0, int k0, int lane) {
  int m  = m0 + (lane & 15);
  int ks = (lane < 16) ? 0 : 16;
  const __bf16* p0 = mat + m * 256 + k0 + ks;
  V16U u;
  u.q[0] = *(const uint4*)p0;
  u.q[1] = *(const uint4*)(p0 + 8);
  return u.v;
}

// B fragment (32x16 of W^T) from LDS weights stored row-major (n,k), stride 256.
__device__ __forceinline__ v16bf b_frag_lds(const __bf16* w, int n0, int k0, int lane) {
  int n  = n0 + (lane & 15);
  int kb = k0 + ((lane < 16) ? 0 : 16);
  const __bf16* p = w + n * 256 + kb;
  V16U u;
  u.q[0] = *(const uint4*)p;
  u.q[1] = *(const uint4*)(p + 8);
  return u.v;
}

// B fragment straight from global f32 weights (row-major (n,k), stride 256).
__device__ __forceinline__ v16bf b_frag_gl(const float* w, int n0, int k0, int lane) {
  int n  = n0 + (lane & 15);
  int kb = k0 + ((lane < 16) ? 0 : 16);
  const float4* p = (const float4*)(w + n * 256 + kb);
  v16bf r;
#pragma unroll
  for (int c = 0; c < 4; ++c) {
    float4 f = p[c];
    r[4 * c + 0] = (__bf16)f.x; r[4 * c + 1] = (__bf16)f.y;
    r[4 * c + 2] = (__bf16)f.z; r[4 * c + 3] = (__bf16)f.w;
  }
  return r;
}

// ---------------------------------------------------------------------------
// Weight pre-conversion: wq1, w_o1, wk2, wv2  f32 -> bf16 in workspace.
// 64 blocks x 256 threads; each thread converts one float4 of each matrix.
// ---------------------------------------------------------------------------
__global__ __launch_bounds__(256) void k_prep_w(
    const float* __restrict__ w_qkv1, const float* __restrict__ w_o1,
    const float* __restrict__ w_qkv2, __bf16* __restrict__ dst) {
  const int i = blockIdx.x * 256 + threadIdx.x;    // 0..16383 (float4 index)
  const float* srcs[4] = {w_qkv1, w_o1, w_qkv2 + 65536, w_qkv2 + 131072};
#pragma unroll
  for (int m = 0; m < 4; ++m) {
    float4 f = ((const float4*)srcs[m])[i];
    __bf16* d = dst + m * 65536 + i * 4;
    d[0] = (__bf16)f.x; d[1] = (__bf16)f.y; d[2] = (__bf16)f.z; d[3] = (__bf16)f.w;
  }
}

// ---------------------------------------------------------------------------
// Kernel 0: camera projections. kc = cam@wk1^T+bk1, vc = cam@wv1^T+bv1,
//           q2 = ln2(cam)@wq2^T+bq2.  16 camera tokens per block, 8 waves.
// ---------------------------------------------------------------------------
__global__ __launch_bounds__(256) void k_cam_proj(
    const float* __restrict__ camera, const float* __restrict__ ln2_g,
    const float* __restrict__ ln2_b,  const float* __restrict__ w_qkv1,
    const float* __restrict__ b_qkv1, const float* __restrict__ w_qkv2,
    const float* __restrict__ b_qkv2,
    float* __restrict__ kc, float* __restrict__ vc, float* __restrict__ q2) {
  __shared__ __bf16 ls_raw[16 * 256];
  __shared__ __bf16 ls_ln[16 * 256];
  const int tid = threadIdx.x, lane = tid & 31, wv = tid >> 5;
  const int tb = blockIdx.x * 16;

#pragma unroll
  for (int rr = 0; rr < 2; ++rr) {
    int r = wv * 2 + rr;
    const float* src = camera + (tb + r) * 256;
    float4 f0 = ((const float4*)src)[lane * 2];
    float4 f1 = ((const float4*)src)[lane * 2 + 1];
    float v[8] = {f0.x, f0.y, f0.z, f0.w, f1.x, f1.y, f1.z, f1.w};
    float s = 0.f;
#pragma unroll
    for (int i = 0; i < 8; ++i) s += v[i];
    float mu = wave_sum(s) * (1.f / 256.f);
    float sq = 0.f;
#pragma unroll
    for (int i = 0; i < 8; ++i) { float d = v[i] - mu; sq += d * d; }
    float rstd = rsqrtf(wave_sum(sq) * (1.f / 256.f) + 1e-5f);
#pragma unroll
    for (int i = 0; i < 8; ++i) {
      int c = lane * 8 + i;
      ls_raw[r * 256 + c] = (__bf16)v[i];
      ls_ln[r * 256 + c]  = (__bf16)((v[i] - mu) * rstd * ln2_g[c] + ln2_b[c]);
    }
  }
  __syncthreads();

  const float* wm[3] = {w_qkv1 + 65536, w_qkv1 + 131072, w_qkv2}; // wk1, wv1, wq2
  const float* bm[3] = {b_qkv1 + 256,   b_qkv1 + 512,    b_qkv2};
  float*       om[3] = {kc, vc, q2};
#pragma unroll
  for (int j = 0; j < 6; ++j) {
    int tt  = wv + j * 8;
    int mat = tt >> 4, nt = tt & 15;
    const __bf16* amat = (mat == 2) ? ls_ln : ls_raw;
    v8f acc = {};
#pragma unroll
    for (int ks = 0; ks < 8; ++ks) {
      v16bf A = a_frag(amat, 0, ks * 32, lane);
      v16bf B = b_frag_gl(wm[mat], nt * 16, ks * 32, lane);
      acc = __builtin_amdgcn_wmma_f32_16x16x32_bf16(false, A, false, B,
                                                    (short)0, acc, false, false);
    }
    int col  = nt * 16 + (lane & 15);
    int rOff = (lane < 16) ? 0 : 8;
    float bias = bm[mat][col];
    float* out = om[mat];
#pragma unroll
    for (int i = 0; i < 8; ++i)
      out[(tb + rOff + i) * 256 + col] = acc[i] + bias;
  }
}

// ---------------------------------------------------------------------------
// Kernel 1: pose stage.  128 pose tokens per block (8 waves x 16 rows).
// TDM(wq1) overlapped with LN1 -> WMMA Q-proj -> TDM(w_o1) overlapped with
// 8-key softmax attention -> WMMA O-proj + residual -> pose_out.
// ---------------------------------------------------------------------------
__global__ __launch_bounds__(256) void k_pose_stage(
    const float* __restrict__ pose,  const float* __restrict__ ln1_g,
    const float* __restrict__ ln1_b, const __bf16* __restrict__ wbf,
    const float* __restrict__ b_qkv1, const float* __restrict__ b_o1,
    const float* __restrict__ kc,    const float* __restrict__ vc,
    float* __restrict__ pose_out) {
  extern __shared__ char smem[];
  __bf16* ls_w  = (__bf16*)(smem);                    // 256x256 bf16 (128 KB, off 0)
  __bf16* ls_a  = (__bf16*)(smem + 131072);           // 128x256 bf16 (qn -> o1)
  __bf16* ls_q  = (__bf16*)(smem + 131072 + 65536);   // 128x256 bf16 (q)
  float*  ls_kc = (float*)(smem + 262144);            // 8x256 f32
  float*  ls_vc = (float*)(smem + 270336);            // 8x256 f32

  const int tid = threadIdx.x, lane = tid & 31, wv = tid >> 5;
  const int bt = blockIdx.x / 6, chunk = blockIdx.x % 6;
  const int rowBase = bt * 768 + chunk * 128;
  const int m0 = wv * 16;

  // Kick off async DMA of wq1(bf16) into ls_w; overlap with kc/vc staging + LN.
  if (wv == 0) tdm_load_128k(wbf, 0u);

  {
    const float* kc_bt = kc + bt * 8 * 256;
    const float* vc_bt = vc + bt * 8 * 256;
#pragma unroll
    for (int i = tid; i < 512; i += 256) {
      ((float4*)ls_kc)[i] = ((const float4*)kc_bt)[i];
      ((float4*)ls_vc)[i] = ((const float4*)vc_bt)[i];
    }
  }
  // LayerNorm(ln1): 16 rows per wave, one row at a time across 32 lanes.
  for (int r = 0; r < 16; ++r) {
    const float* src = pose + (rowBase + m0 + r) * 256;
    float4 f0 = ((const float4*)src)[lane * 2];
    float4 f1 = ((const float4*)src)[lane * 2 + 1];
    float v[8] = {f0.x, f0.y, f0.z, f0.w, f1.x, f1.y, f1.z, f1.w};
    float s = 0.f;
#pragma unroll
    for (int i = 0; i < 8; ++i) s += v[i];
    float mu = wave_sum(s) * (1.f / 256.f);
    float sq = 0.f;
#pragma unroll
    for (int i = 0; i < 8; ++i) { float d = v[i] - mu; sq += d * d; }
    float rstd = rsqrtf(wave_sum(sq) * (1.f / 256.f) + 1e-5f);
    __bf16* dst = ls_a + (m0 + r) * 256 + lane * 8;
#pragma unroll
    for (int i = 0; i < 8; ++i) {
      int c = lane * 8 + i;
      dst[i] = (__bf16)((v[i] - mu) * rstd * ln1_g[c] + ln1_b[c]);
    }
  }
  if (wv == 0) __builtin_amdgcn_s_wait_tensorcnt(0);
  __syncthreads();

  // ---- GEMM Q = ln1(pose) @ wq1^T + bq1 -> ls_q (bf16) ----
  {
    v8f acc[16];
#pragma unroll
    for (int nt = 0; nt < 16; ++nt) acc[nt] = (v8f){};
    for (int ks = 0; ks < 8; ++ks) {
      v16bf A = a_frag(ls_a, m0, ks * 32, lane);
#pragma unroll
      for (int nt = 0; nt < 16; ++nt) {
        v16bf B = b_frag_lds(ls_w, nt * 16, ks * 32, lane);
        acc[nt] = __builtin_amdgcn_wmma_f32_16x16x32_bf16(false, A, false, B,
                                                          (short)0, acc[nt], false, false);
      }
    }
    int rOff = (lane < 16) ? 0 : 8;
#pragma unroll
    for (int nt = 0; nt < 16; ++nt) {
      int col = nt * 16 + (lane & 15);
      float bias = b_qkv1[col];
#pragma unroll
      for (int i = 0; i < 8; ++i)
        ls_q[(m0 + rOff + i) * 256 + col] = (__bf16)(acc[nt][i] + bias);
    }
  }
  __syncthreads();

  // Async DMA of w_o1(bf16) into ls_w while attention runs.
  if (wv == 0) tdm_load_128k(wbf + 65536, 0u);

  // ---- Attention: each (token,head) pair attends over 8 camera keys ----
#pragma unroll
  for (int it = 0; it < 4; ++it) {
    int p   = tid + it * 256;
    int row = p >> 3, h = p & 7;
    const __bf16* qrow = ls_q + row * 256 + h * 32;
    float qv[32];
#pragma unroll
    for (int d = 0; d < 32; ++d) qv[d] = (float)qrow[d];
    float s[8];
    float mx = -3.0e38f;
#pragma unroll
    for (int k = 0; k < 8; ++k) {
      const float* kp = ls_kc + k * 256 + h * 32;
      float a = 0.f;
#pragma unroll
      for (int d = 0; d < 32; ++d) a += qv[d] * kp[d];
      s[k] = a * ATT_SCALE;
      mx = fmaxf(mx, s[k]);
    }
    float ssum = 0.f;
#pragma unroll
    for (int k = 0; k < 8; ++k) { s[k] = __expf(s[k] - mx); ssum += s[k]; }
    float inv = 1.f / ssum;
    float o[32];
#pragma unroll
    for (int d = 0; d < 32; ++d) o[d] = 0.f;
#pragma unroll
    for (int k = 0; k < 8; ++k) {
      float a = s[k] * inv;
      const float* vp = ls_vc + k * 256 + h * 32;
#pragma unroll
      for (int d = 0; d < 32; ++d) o[d] += a * vp[d];
    }
    __bf16* orow = ls_a + row * 256 + h * 32;
#pragma unroll
    for (int d = 0; d < 32; ++d) orow[d] = (__bf16)o[d];
  }
  if (wv == 0) __builtin_amdgcn_s_wait_tensorcnt(0);
  __syncthreads();

  // ---- GEMM O = o1 @ w_o1^T + b_o1 + pose residual -> pose_out ----
  {
    v8f acc[16];
#pragma unroll
    for (int nt = 0; nt < 16; ++nt) acc[nt] = (v8f){};
    for (int ks = 0; ks < 8; ++ks) {
      v16bf A = a_frag(ls_a, m0, ks * 32, lane);
#pragma unroll
      for (int nt = 0; nt < 16; ++nt) {
        v16bf B = b_frag_lds(ls_w, nt * 16, ks * 32, lane);
        acc[nt] = __builtin_amdgcn_wmma_f32_16x16x32_bf16(false, A, false, B,
                                                          (short)0, acc[nt], false, false);
      }
    }
    int rOff = (lane < 16) ? 0 : 8;
#pragma unroll
    for (int nt = 0; nt < 16; ++nt) {
      int col = nt * 16 + (lane & 15);
      float bias = b_o1[col];
#pragma unroll
      for (int i = 0; i < 8; ++i) {
        int gidx = (rowBase + m0 + rOff + i) * 256 + col;
        pose_out[gidx] = acc[nt][i] + bias + pose[gidx];
      }
    }
  }
}

// ---------------------------------------------------------------------------
// Kernel 2: camera stage.  One block per camera token (b,t,k).
// TDM(wk2) -> WMMA K2-proj -> TDM(wv2) overlapped with softmax over 96 keys
// -> WMMA V2-proj -> o2 accumulation -> output GEMV + residual -> cam_out.
// ---------------------------------------------------------------------------
__global__ __launch_bounds__(256) void k_cam_stage(
    const float* __restrict__ camera, const __bf16* __restrict__ wbf,
    const float* __restrict__ b_qkv2, const float* __restrict__ w_o2,
    const float* __restrict__ b_o2,   const float* __restrict__ q2buf,
    const float* __restrict__ pose_out, float* __restrict__ cam_out) {
  extern __shared__ char smem[];
  __bf16* ls_w  = (__bf16*)(smem);            // 256x256 bf16 (128 KB, off 0)
  __bf16* ls_kv = (__bf16*)(smem + 131072);   // 96x256 bf16
  __bf16* ls_k2 = (__bf16*)(smem + 180224);   // 96x256 bf16
  __bf16* ls_v2 = (__bf16*)(smem + 229376);   // 96x256 bf16
  float*  ls_p  = (float*)(smem + 278528);    // 8 x 96 probs
  float*  ls_o2 = (float*)(smem + 281600);    // 256 f32

  const int tid = threadIdx.x, lane = tid & 31, wv = tid >> 5;
  const int btk = blockIdx.x;

  // Async DMA of wk2(bf16) into ls_w; overlap with kv staging.
  if (wv == 0) tdm_load_128k(wbf + 131072, 0u);

  const float* kvsrc = pose_out + btk * 96 * 256;
  for (int i = tid; i < 6144; i += 256) {
    float4 f = ((const float4*)kvsrc)[i];
    __bf16* d = ls_kv + i * 4;
    d[0] = (__bf16)f.x; d[1] = (__bf16)f.y; d[2] = (__bf16)f.z; d[3] = (__bf16)f.w;
  }
  if (wv == 0) __builtin_amdgcn_s_wait_tensorcnt(0);
  __syncthreads();

  // ---- GEMM K2 = kv @ wk2^T + bk2 -> ls_k2 ----
  for (int j = 0; j < 12; ++j) {
    int tt = wv + j * 8;
    int mt = tt >> 4, nt = tt & 15;
    v8f acc = {};
#pragma unroll
    for (int ks = 0; ks < 8; ++ks) {
      v16bf A = a_frag(ls_kv, mt * 16, ks * 32, lane);
      v16bf B = b_frag_lds(ls_w, nt * 16, ks * 32, lane);
      acc = __builtin_amdgcn_wmma_f32_16x16x32_bf16(false, A, false, B,
                                                    (short)0, acc, false, false);
    }
    int col  = nt * 16 + (lane & 15);
    int rOff = (lane < 16) ? 0 : 8;
    float bias = b_qkv2[256 + col];
#pragma unroll
    for (int i = 0; i < 8; ++i)
      ls_k2[(mt * 16 + rOff + i) * 256 + col] = (__bf16)(acc[i] + bias);
  }
  __syncthreads();

  // Async DMA of wv2(bf16) into ls_w while scores+softmax run on ls_k2.
  if (wv == 0) tdm_load_128k(wbf + 196608, 0u);

  // ---- Attention scores + softmax: wave = head, 96 pose keys ----
  {
    const int h = wv;
    const float* q2h = q2buf + btk * 256 + h * 32;
    float qv[32];
#pragma unroll
    for (int d = 0; d < 32; ++d) qv[d] = q2h[d];
    float s[3];
    float mx = -3.0e38f;
#pragma unroll
    for (int c = 0; c < 3; ++c) {
      int n = lane + c * 32;
      const __bf16* kp = ls_k2 + n * 256 + h * 32;
      float a = 0.f;
#pragma unroll
      for (int d = 0; d < 32; ++d) a += qv[d] * (float)kp[d];
      s[c] = a * ATT_SCALE;
      mx = fmaxf(mx, s[c]);
    }
    mx = wave_max(mx);
    float sum = 0.f;
#pragma unroll
    for (int c = 0; c < 3; ++c) { s[c] = __expf(s[c] - mx); sum += s[c]; }
    sum = wave_sum(sum);
    float inv = 1.f / sum;
#pragma unroll
    for (int c = 0; c < 3; ++c) ls_p[h * 96 + lane + c * 32] = s[c] * inv;
  }
  if (wv == 0) __builtin_amdgcn_s_wait_tensorcnt(0);
  __syncthreads();

  // ---- GEMM V2 = kv @ wv2^T + bv2 -> ls_v2 ----
  for (int j = 0; j < 12; ++j) {
    int tt = wv + j * 8;
    int mt = tt >> 4, nt = tt & 15;
    v8f acc = {};
#pragma unroll
    for (int ks = 0; ks < 8; ++ks) {
      v16bf A = a_frag(ls_kv, mt * 16, ks * 32, lane);
      v16bf B = b_frag_lds(ls_w, nt * 16, ks * 32, lane);
      acc = __builtin_amdgcn_wmma_f32_16x16x32_bf16(false, A, false, B,
                                                    (short)0, acc, false, false);
    }
    int col  = nt * 16 + (lane & 15);
    int rOff = (lane < 16) ? 0 : 8;
    float bias = b_qkv2[512 + col];
#pragma unroll
    for (int i = 0; i < 8; ++i)
      ls_v2[(mt * 16 + rOff + i) * 256 + col] = (__bf16)(acc[i] + bias);
  }
  __syncthreads();

  // ---- Weighted sum of values: wave = head, lane = dh element ----
  {
    const int h = wv, d = lane;
    float o = 0.f;
    for (int n = 0; n < 96; ++n)
      o += ls_p[h * 96 + n] * (float)ls_v2[n * 256 + h * 32 + d];
    ls_o2[h * 32 + d] = o;
  }
  __syncthreads();

  // ---- Output GEMV: cam_out = camera + o2 @ w_o2^T + b_o2 ----
  {
    const int n = tid;
    const float4* w4 = (const float4*)(w_o2 + n * 256);
    const float4* o4 = (const float4*)ls_o2;
    float r = b_o2[n];
#pragma unroll 8
    for (int k = 0; k < 64; ++k) {
      float4 a = o4[k];
      float4 b = w4[k];
      r += a.x * b.x + a.y * b.y + a.z * b.z + a.w * b.w;
    }
    cam_out[btk * 256 + n] = camera[btk * 256 + n] + r;
  }
}

// ---------------------------------------------------------------------------
extern "C" void kernel_launch(void* const* d_in, const int* in_sizes, int n_in,
                              void* d_out, int out_size, void* d_ws, size_t ws_size,
                              hipStream_t stream) {
  (void)in_sizes; (void)n_in; (void)out_size; (void)ws_size;
  const float* pose   = (const float*)d_in[0];
  const float* camera = (const float*)d_in[1];
  const float* ln1_g  = (const float*)d_in[2];
  const float* ln1_b  = (const float*)d_in[3];
  const float* w_qkv1 = (const float*)d_in[4];
  const float* b_qkv1 = (const float*)d_in[5];
  const float* w_o1   = (const float*)d_in[6];
  const float* b_o1   = (const float*)d_in[7];
  const float* ln2_g  = (const float*)d_in[8];
  const float* ln2_b  = (const float*)d_in[9];
  const float* w_qkv2 = (const float*)d_in[10];
  const float* b_qkv2 = (const float*)d_in[11];
  const float* w_o2   = (const float*)d_in[12];
  const float* b_o2   = (const float*)d_in[13];

  float* pose_out = (float*)d_out;
  float* cam_out  = (float*)d_out + POSE_ELEMS;

  float* ws = (float*)d_ws;
  float* kc = ws;                    // 2048 x 256
  float* vc = ws + CAM_ELEMS;        // 2048 x 256
  float* q2 = ws + 2 * CAM_ELEMS;    // 2048 x 256
  __bf16* wbf = (__bf16*)(ws + WBF_FOFF);  // wq1 | w_o1 | wk2 | wv2 (bf16)

  k_prep_w<<<64, 256, 0, stream>>>(w_qkv1, w_o1, w_qkv2, wbf);
  k_cam_proj<<<128, 256, 0, stream>>>(camera, ln2_g, ln2_b, w_qkv1, b_qkv1,
                                      w_qkv2, b_qkv2, kc, vc, q2);
  k_pose_stage<<<1536, 256, 278528, stream>>>(pose, ln1_g, ln1_b, wbf, b_qkv1,
                                              b_o1, kc, vc, pose_out);
  k_cam_stage<<<2048, 256, 282624, stream>>>(camera, wbf, b_qkv2, w_o2, b_o2,
                                             q2, pose_out, cam_out);
}
